// PureTorchSVDQInt4LinearBF16_1898375544896
// MI455X (gfx1250) — 8-bit matrix operand order repaired and checked by reading, compiled, not run on hardware
//
#include <hip/hip_runtime.h>
#include <hip/hip_bf16.h>
#include <stdint.h>

#define N_TOKENS 4096
#define IN_F     4096
#define OUT_F    4096
#define RANK     32
#define GS       64
#define NG       64

typedef int   v8i __attribute__((ext_vector_type(8)));
typedef int   v4i_g __attribute__((vector_size(16)));

#define AS1 __attribute__((address_space(1)))
#define AS3 __attribute__((address_space(3)))

#if defined(__gfx1250__) && __has_builtin(__builtin_amdgcn_global_load_async_to_lds_b128) && __has_builtin(__builtin_amdgcn_s_wait_asynccnt)
#define USE_ASYNC_LDS 1
#else
#define USE_ASYNC_LDS 0
#endif

#define QX_OFF   0u
#define W8_OFF   (16u * 1024u * 1024u)
#define LR_OFF   (32u * 1024u * 1024u)
#define ASC_OFF  (LR_OFF + (N_TOKENS * RANK * 4u))

__global__ __launch_bounds__(256) void quantize_kernel(const float* __restrict__ x,
                                                       int8_t* __restrict__ qx,
                                                       float* __restrict__ ascale) {
    const int lane = threadIdx.x & 31;
    const int wid  = blockIdx.x * 8 + (threadIdx.x >> 5);
    const int tok  = wid >> 6;
    const int g    = wid & 63;
    const int base = tok * IN_F + g * GS + lane * 2;

    const float2 v = *(const float2*)(x + base);
    float m = fmaxf(fabsf(v.x), fabsf(v.y));
    #pragma unroll
    for (int off = 16; off > 0; off >>= 1)
        m = fmaxf(m, __shfl_xor(m, off, 32));

    const float asc = (m > 0.0f) ? (m * (1.0f / 127.0f)) : 1.0f;
    const float inv = 1.0f / asc;
    float q0 = rintf(v.x * inv);
    float q1 = rintf(v.y * inv);
    q0 = fminf(fmaxf(q0, -127.0f), 127.0f);
    q1 = fminf(fmaxf(q1, -127.0f), 127.0f);

    char2 packed;
    packed.x = (char)(int)q0;
    packed.y = (char)(int)q1;
    *(char2*)(qx + base) = packed;
    if (lane == 0) ascale[tok * NG + g] = asc;
}

__global__ __launch_bounds__(256) void packw_kernel(const int* __restrict__ qw,
                                                    uint32_t* __restrict__ w8) {
    const int gid = blockIdx.x * 256 + threadIdx.x;
    const int4 q  = *(const int4*)(qw + (size_t)gid * 4);
    uint32_t p = (uint32_t)(q.x & 0xFF) |
                 ((uint32_t)(q.y & 0xFF) << 8) |
                 ((uint32_t)(q.z & 0xFF) << 16) |
                 ((uint32_t)(q.w & 0xFF) << 24);
    w8[gid] = p;
}

__global__ __launch_bounds__(256) void lowrank_kernel(const float* __restrict__ x,
                                                      const float* __restrict__ pd,
                                                      float* __restrict__ lr) {
    const int lane = threadIdx.x & 31;
    const int tok  = blockIdx.x * 8 + (threadIdx.x >> 5);
    const float* xr = x + (size_t)tok * IN_F;
    float acc = 0.0f;
    #pragma unroll 4
    for (int k = 0; k < IN_F; ++k)
        acc = fmaf(xr[k], pd[k * RANK + lane], acc);
    lr[tok * RANK + lane] = acc;
}

#define TSTRIDE 80
#define NBUF    3
#define BN      128

__global__ __launch_bounds__(256) void gemm_kernel(const int8_t* __restrict__ qx,
                                                   const int8_t* __restrict__ w8,
                                                   const float* __restrict__ ascale,
                                                   const float* __restrict__ wscale,
                                                   const float* __restrict__ bias,
                                                   const float* __restrict__ lr,
                                                   const float* __restrict__ pu,
                                                   float* __restrict__ out) {
    __shared__ __align__(16) unsigned char Asm[NBUF][64 * TSTRIDE];
    __shared__ __align__(16) unsigned char Bsm[NBUF][BN * TSTRIDE];
    __shared__ float aS[NG * 64];
    __shared__ float wS[NG * BN];
    __shared__ float lrS[64 * RANK];

    const int tid   = threadIdx.x;
    const int lane  = tid & 31;
    const int wave  = tid >> 5;
    const int l15   = lane & 15;
    const int half  = lane >> 4;
    const int subM  = wave & 3;
    const int subN0 = (wave >> 2) * 4;

    const int tokBase = blockIdx.y * 64;
    const int outBase = blockIdx.x * BN;

    const int row = tid >> 2;
    const int seg = (tid & 3) * 16;
    const int8_t* gArow  = qx + (size_t)(tokBase + row) * IN_F + seg;
    const int8_t* gBrow0 = w8 + (size_t)(outBase + row) * IN_F + seg;
    const int8_t* gBrow1 = w8 + (size_t)(outBase + row + 64) * IN_F + seg;

    for (int i = tid; i < NG * 64; i += 256) {
        const int t = i >> 6, g = i & 63;
        aS[g * 64 + t] = ascale[(size_t)(tokBase + t) * NG + g];
    }
    for (int i = tid; i < NG * BN; i += 256) {
        const int o = i >> 6, g = i & 63;
        wS[g * BN + o] = wscale[(size_t)(outBase + o) * NG + g];
    }
    for (int i = tid; i < 64 * RANK; i += 256)
        lrS[i] = lr[(size_t)tokBase * RANK + i];

    float acc[4][8];
    #pragma unroll
    for (int s = 0; s < 4; ++s)
        #pragma unroll
        for (int r = 0; r < 8; ++r) acc[s][r] = 0.0f;

    union VA { v8i v; uint2 u2[4]; } Af;
    union VB { v8i v; uint2 u2[4]; } Bf;
    union VC { v8i v; int i[8]; } Cf;
    const v8i zeroi = {};

#if USE_ASYNC_LDS
    auto issue_tile = [&](int g, int buf) {
        __builtin_amdgcn_global_load_async_to_lds_b128(
            (AS1 v4i_g*)(gArow + (size_t)g * GS),
            (AS3 v4i_g*)(&Asm[buf][row * TSTRIDE + seg]), 0, 0);
        __builtin_amdgcn_global_load_async_to_lds_b128(
            (AS1 v4i_g*)(gBrow0 + (size_t)g * GS),
            (AS3 v4i_g*)(&Bsm[buf][row * TSTRIDE + seg]), 0, 0);
        __builtin_amdgcn_global_load_async_to_lds_b128(
            (AS1 v4i_g*)(gBrow1 + (size_t)g * GS),
            (AS3 v4i_g*)(&Bsm[buf][(row + 64) * TSTRIDE + seg]), 0, 0);
    };
    issue_tile(0, 0);
    issue_tile(1, 1);
#endif

    for (int g = 0; g < NG; ++g) {
        const int cur = g % NBUF;
#if USE_ASYNC_LDS
        if (g + 1 < NG) __builtin_amdgcn_s_wait_asynccnt(3);
        else            __builtin_amdgcn_s_wait_asynccnt(0);
        __syncthreads();
        if (g + 2 < NG) issue_tile(g + 2, (g + 2) % NBUF);
#else
        *(uint4*)(&Asm[cur][row * TSTRIDE + seg]) =
            *(const uint4*)(gArow + (size_t)g * GS);
        *(uint4*)(&Bsm[cur][row * TSTRIDE + seg]) =
            *(const uint4*)(gBrow0 + (size_t)g * GS);
        *(uint4*)(&Bsm[cur][(row + 64) * TSTRIDE + seg]) =
            *(const uint4*)(gBrow1 + (size_t)g * GS);
        __syncthreads();
#endif

        const unsigned char* arow = &Asm[cur][(subM * 16 + l15) * TSTRIDE + half * 8];
        Af.u2[0] = *(const uint2*)(arow + 0);
        Af.u2[1] = *(const uint2*)(arow + 16);
        Af.u2[2] = *(const uint2*)(arow + 32);
        Af.u2[3] = *(const uint2*)(arow + 48);

        const float* ap = &aS[g * 64 + subM * 16 + half * 8];
        float a8[8];
        #pragma unroll
        for (int r = 0; r < 8; ++r) a8[r] = ap[r];

        #pragma unroll
        for (int s = 0; s < 4; ++s) {
            const int nLoc = (subN0 + s) * 16;
            const unsigned char* brow = &Bsm[cur][(nLoc + l15) * TSTRIDE + half * 8];
            Bf.u2[0] = *(const uint2*)(brow + 0);
            Bf.u2[1] = *(const uint2*)(brow + 16);
            Bf.u2[2] = *(const uint2*)(brow + 32);
            Bf.u2[3] = *(const uint2*)(brow + 48);

            Cf.v = __builtin_amdgcn_wmma_i32_16x16x64_iu8(
                true, Af.v, true, Bf.v, zeroi, false, false);

            const float wsv = wS[g * BN + nLoc + l15];
            #pragma unroll
            for (int r = 0; r < 8; ++r)
                acc[s][r] = fmaf((float)Cf.i[r], a8[r] * wsv, acc[s][r]);
        }
#if !USE_ASYNC_LDS
        __syncthreads();
#endif
    }

    const int mBase = subM * 16 + half * 8;
    #pragma unroll
    for (int s = 0; s < 4; ++s) {
        const int n = outBase + (subN0 + s) * 16 + l15;
        const float b = bias[n];
        float puv[RANK];
        #pragma unroll
        for (int j = 0; j < RANK; ++j) puv[j] = pu[(size_t)j * OUT_F + n];
        #pragma unroll
        for (int r = 0; r < 8; ++r) {
            float v = acc[s][r] + b;
            const float* lrow = &lrS[(mBase + r) * RANK];
            #pragma unroll
            for (int j = 0; j < RANK; ++j) v = fmaf(lrow[j], puv[j], v);
            out[(size_t)(tokBase + mBase + r) * OUT_F + n] = v;
        }
    }
}

extern "C" void kernel_launch(void* const* d_in, const int* in_sizes, int n_in,
                              void* d_out, int out_size, void* d_ws, size_t ws_size,
                              hipStream_t stream) {
    (void)in_sizes; (void)n_in; (void)out_size; (void)ws_size;

    const float* x      = (const float*)d_in[0];
    const int*   qw     = (const int*)  d_in[1];
    const float* wscale = (const float*)d_in[2];
    const float* bias   = (const float*)d_in[3];
    const float* pd     = (const float*)d_in[4];
    const float* pu     = (const float*)d_in[5];
    float* out = (float*)d_out;

    char* ws = (char*)d_ws;
    int8_t*   qx  = (int8_t*)(ws + QX_OFF);
    int8_t*   w8  = (int8_t*)(ws + W8_OFF);
    float*    lrv = (float*) (ws + LR_OFF);
    float*    asc = (float*) (ws + ASC_OFF);

    quantize_kernel<<<(N_TOKENS * NG) / 8, 256, 0, stream>>>(x, qx, asc);
    packw_kernel<<<(OUT_F * IN_F) / (256 * 4), 256, 0, stream>>>(qw, (uint32_t*)w8);
    lowrank_kernel<<<N_TOKENS / 8, 256, 0, stream>>>(x, pd, lrv);
    dim3 grid(OUT_F / BN, N_TOKENS / 64);
    gemm_kernel<<<grid, 256, 0, stream>>>(qx, w8, asc, wscale, bias, lrv, pu, out);
}
